// MambaBlock_18777597018326
// MI455X (gfx1250) — compile-verified, matrix operand order checked by reading, not run on hardware
//
#include <hip/hip_runtime.h>

// ---------------------------------------------------------------------------
// Mamba block for MI455X (gfx1250, wave32).
//  - GEMMs: v_wmma_f32_16x16x32_bf16, 256x128 block tile, 64x64 wave tile
//    (16 WMMA / 16 ds_load_b128 per wave per K-step), double-buffered LDS
//    filled with GLOBAL_LOAD_ASYNC_TO_LDS_B128 (ASYNCcnt) for overlap.
//  - Selective scan: 1 channel/lane, h[16] in registers, B/C broadcast via
//    double-buffered LDS (1 barrier per timestep).
// ---------------------------------------------------------------------------

typedef __attribute__((ext_vector_type(16))) __bf16 v16bf;
typedef __attribute__((ext_vector_type(8)))  __bf16 v8bf;
typedef __attribute__((ext_vector_type(8)))  float  v8f;
typedef unsigned short u16;
typedef unsigned int   u32;
typedef __attribute__((address_space(3))) char lds_char;

#define D_MODEL 1024
#define D_STATE 16
#define DT_RANK 64
#define D_INNER 2048
#define SEQ_L   2048
#define BATCH   2
#define NTOK    (BATCH * SEQ_L)          // 4096 token rows
#define XPROJ_N (DT_RANK + 2 * D_STATE)  // 96
#define XPROJ_NPAD 128                   // padded to a full BN tile

__device__ __forceinline__ u16 f2bf(float f) {
  u32 u = __float_as_uint(f);
  u32 r = u + 0x7fffu + ((u >> 16) & 1u);   // round-to-nearest-even
  return (u16)(r >> 16);
}

// LDS byte offset of a generic pointer known to point into LDS
__device__ __forceinline__ unsigned lds_off(const void* p) {
  return (unsigned)(size_t)(const lds_char*)p;
}

// 16B-per-lane async global->LDS copy (VGLOBAL GLOBAL_LOAD_ASYNC_TO_LDS_B128)
__device__ __forceinline__ void async_load_b128(const void* gp, void* lp) {
  unsigned l = lds_off(lp);
  asm volatile("global_load_async_to_lds_b128 %0, %1, off"
               :
               : "v"(l), "v"(gp)
               : "memory");
}

__device__ __forceinline__ void wait_async0() {
  asm volatile("s_wait_asynccnt 0x0" ::: "memory");
}

// ---------------------------------------------------------------------------
// Strided f32 -> bf16 convert; rows in [rows_src, rows_dst) are zero-filled
// (used to pad W_x to a full 128-row tile so GEMM loads are unguarded).
// ---------------------------------------------------------------------------
__global__ void k_f32_to_bf16(const float* __restrict__ src, u16* __restrict__ dst,
                              int rows_src, int rows_dst, int cols,
                              int sstride, int dstride) {
  int n = rows_dst * cols;
  for (int i = blockIdx.x * blockDim.x + threadIdx.x; i < n;
       i += gridDim.x * blockDim.x) {
    int r = i / cols, c = i - r * cols;
    dst[(size_t)r * dstride + c] =
        (r < rows_src) ? f2bf(src[(size_t)r * sstride + c]) : (u16)0;
  }
}

// ---------------------------------------------------------------------------
// LayerNorm over D_MODEL=1024, one row per block, emits bf16 for WMMA A-matrix
// ---------------------------------------------------------------------------
__global__ __launch_bounds__(256)
void k_layernorm_bf16(const float* __restrict__ x, const float* __restrict__ w,
                      const float* __restrict__ b, u16* __restrict__ out) {
  __shared__ float red[256];
  int row = blockIdx.x;
  int t = threadIdx.x;
  const float* xr = x + (size_t)row * D_MODEL;

  float v[4];
  float s = 0.f;
#pragma unroll
  for (int i = 0; i < 4; ++i) { v[i] = xr[t + i * 256]; s += v[i]; }
  red[t] = s; __syncthreads();
  for (int off = 128; off > 0; off >>= 1) {
    if (t < off) red[t] += red[t + off];
    __syncthreads();
  }
  float mu = red[0] * (1.0f / D_MODEL);
  __syncthreads();

  float s2 = 0.f;
#pragma unroll
  for (int i = 0; i < 4; ++i) { float d = v[i] - mu; s2 += d * d; }
  red[t] = s2; __syncthreads();
  for (int off = 128; off > 0; off >>= 1) {
    if (t < off) red[t] += red[t + off];
    __syncthreads();
  }
  float rstd = rsqrtf(red[0] * (1.0f / D_MODEL) + 1e-5f);

#pragma unroll
  for (int i = 0; i < 4; ++i) {
    int col = t + i * 256;
    out[(size_t)row * D_MODEL + col] = f2bf((v[i] - mu) * rstd * w[col] + b[col]);
  }
}

// ---------------------------------------------------------------------------
// NT GEMM: C[M,N] = A[M,K](bf16) * B[N,K](bf16)^T (+ optional f32 addend)
// Requirements: M % 256 == 0, K % 32 == 0, B has >= ceil(N/128)*128 rows.
// ---------------------------------------------------------------------------
#define BM 256
#define BN 128
#define BK 32
#define LDT 40   // padded LDS row stride in halves (80B, 16B aligned)

__device__ __forceinline__ void issue_tile_loads(
    const u16* __restrict__ A, int lda,
    const u16* __restrict__ B, int ldb,
    int m0, int n0, int k0, int t,
    __bf16 (*At)[LDT], __bf16 (*Bt)[LDT]) {
  // A tile: 256 rows x 32 halves; thread t owns row t (4 x b128)
  const u16* ga = A + (size_t)(m0 + t) * lda + k0;
  char* la = (char*)&At[t][0];
#pragma unroll
  for (int c = 0; c < 4; ++c)
    async_load_b128(ga + c * 8, la + c * 16);
  // B tile: 128 rows x 32 halves; thread t owns half-row (2 x b128)
  int br = t >> 1;
  int lc = (t & 1) * 16;
  const u16* gb = B + (size_t)(n0 + br) * ldb + k0 + lc;
  char* lb = (char*)&Bt[br][lc];
  async_load_b128(gb, lb);
  async_load_b128(gb + 8, lb + 16);
}

__global__ __launch_bounds__(256)
void k_gemm_bf16_nt(const u16* __restrict__ A, int lda,
                    const u16* __restrict__ B, int ldb,
                    float* __restrict__ C, int ldc,
                    int M, int N, int K,
                    const float* __restrict__ addend) {
  __shared__ __align__(16) __bf16 At[2][BM][LDT];
  __shared__ __align__(16) __bf16 Bt[2][BN][LDT];

  int t    = threadIdx.x;
  int lane = t & 31;
  int wave = t >> 5;
  int wm = (wave & 3) * 64;   // 4 waves along M  -> 256
  int wn = (wave >> 2) * 64;  // 2 waves along N  -> 128
  int m0 = blockIdx.y * BM;
  int n0 = blockIdx.x * BN;

  v8f acc[4][4] = {};

  int nk = K / BK;

  // prologue: tile 0 into buffer 0
  issue_tile_loads(A, lda, B, ldb, m0, n0, 0, t, At[0], Bt[0]);
  wait_async0();
  __syncthreads();

  int r    = lane & 15;
  int koff = (lane >> 4) * 8;   // lanes 0-15: K 0-7/16-23 ; 16-31: K 8-15/24-31

  for (int kk = 0; kk < nk; ++kk) {
    int cur = kk & 1;
    if (kk + 1 < nk) {
      // overlap: async-fill the other buffer while we compute on this one
      issue_tile_loads(A, lda, B, ldb, m0, n0, (kk + 1) * BK, t,
                       At[cur ^ 1], Bt[cur ^ 1]);
      if (kk + 2 < nk)  // L2 warm-up two tiles ahead
        __builtin_prefetch(A + (size_t)(m0 + t) * lda + (kk + 2) * BK, 0, 3);
    }

    v16bf af[4], bf[4];
#pragma unroll
    for (int i = 0; i < 4; ++i) {
      const __bf16* p = &At[cur][wm + i * 16 + r][koff];
      v8bf lo = *(const v8bf*)p;
      v8bf hi = *(const v8bf*)(p + 16);
      af[i] = __builtin_shufflevector(lo, hi, 0,1,2,3,4,5,6,7,8,9,10,11,12,13,14,15);
    }
#pragma unroll
    for (int j = 0; j < 4; ++j) {
      const __bf16* p = &Bt[cur][wn + j * 16 + r][koff];
      v8bf lo = *(const v8bf*)p;
      v8bf hi = *(const v8bf*)(p + 16);
      bf[j] = __builtin_shufflevector(lo, hi, 0,1,2,3,4,5,6,7,8,9,10,11,12,13,14,15);
    }
#pragma unroll
    for (int i = 0; i < 4; ++i)
#pragma unroll
      for (int j = 0; j < 4; ++j)
        acc[i][j] = __builtin_amdgcn_wmma_f32_16x16x32_bf16(
            false, af[i], false, bf[j], (short)0, acc[i][j], false, false);

    wait_async0();     // own async stores to LDS done before signaling
    __syncthreads();   // everyone done reading cur + writing next
  }

  // C/D layout: VGPR v, lanes 0-15 -> M=v ; lanes 16-31 -> M=v+8 ; N = lane%16
  int hrow = (lane >> 4) * 8;
  int colb = lane & 15;
#pragma unroll
  for (int i = 0; i < 4; ++i) {
#pragma unroll
    for (int j = 0; j < 4; ++j) {
      int col = n0 + wn + j * 16 + colb;
      if (col >= N) continue;
#pragma unroll
      for (int v = 0; v < 8; ++v) {
        int row = m0 + wm + i * 16 + hrow + v;
        if (row < M) {
          float val = acc[i][j][v];
          if (addend) val += addend[(size_t)row * ldc + col];
          C[(size_t)row * ldc + col] = val;
        }
      }
    }
  }
}

// ---------------------------------------------------------------------------
// Causal depthwise conv1d (D_CONV=4) + SiLU; emits f32 (scan) + bf16 (x_proj)
// u lives in xz rows with stride 2*D_INNER (first half of in_proj output)
// ---------------------------------------------------------------------------
__global__ void k_conv_silu(const float* __restrict__ xz, const float* __restrict__ cw,
                            const float* __restrict__ cb, float* __restrict__ ucf,
                            u16* __restrict__ ucb) {
  int n = NTOK * D_INNER;
  for (int i = blockIdx.x * blockDim.x + threadIdx.x; i < n;
       i += gridDim.x * blockDim.x) {
    int row  = i >> 11;          // token index b*L + t
    int d    = i & (D_INNER - 1);
    int tpos = row & (SEQ_L - 1);
    float s = cb[d];
#pragma unroll
    for (int j = 0; j < 4; ++j) {
      int tt = tpos + j - 3;
      if (tt >= 0)
        s += cw[d * 4 + j] * xz[(size_t)(row + j - 3) * (2 * D_INNER) + d];
    }
    float r = s / (1.f + __expf(-s));   // SiLU
    ucf[i] = r;
    ucb[i] = f2bf(r);
  }
}

// ---------------------------------------------------------------------------
// Selective scan, fused: softplus(dt)+bias, h-recurrence, y=h.C, +D*u,
// *silu(z), bf16 cast for out_proj. One channel per lane, serial over L.
// B_t/C_t (32 floats, shared per batch) double-buffered: 1 barrier/step.
// ---------------------------------------------------------------------------
__global__ __launch_bounds__(256)
void k_scan(const float* __restrict__ dpre, const float* __restrict__ xdbl,
            const float* __restrict__ ucf,  const float* __restrict__ xz,
            const float* __restrict__ A_log, const float* __restrict__ bdt_p,
            const float* __restrict__ Dp,   u16* __restrict__ ybf) {
  __shared__ float bc[2][2 * D_STATE];
  const int blocks_per_batch = D_INNER / 256;   // 8
  int b = blockIdx.x / blocks_per_batch;
  int d = (blockIdx.x % blocks_per_batch) * 256 + threadIdx.x;
  int tid = threadIdx.x;

  float A[D_STATE], h[D_STATE];
#pragma unroll
  for (int s = 0; s < D_STATE; ++s) {
    A[s] = -__expf(A_log[d * D_STATE + s]);
    h[s] = 0.f;
  }
  float bdt = bdt_p[d];
  float Dv  = Dp[d];

  if (tid < 2 * D_STATE)
    bc[0][tid] = xdbl[(size_t)(b * SEQ_L) * XPROJ_N + DT_RANK + tid];
  __syncthreads();

  for (int tpos = 0; tpos < SEQ_L; ++tpos) {
    int row = b * SEQ_L + tpos;
    int cur = tpos & 1;
    if (tid < 2 * D_STATE && tpos + 1 < SEQ_L)
      bc[cur ^ 1][tid] = xdbl[(size_t)(row + 1) * XPROJ_N + DT_RANK + tid];

    float dp    = dpre[(size_t)row * D_INNER + d] + bdt;
    float delta = (dp > 20.f) ? dp : log1pf(__expf(dp));   // softplus
    float u     = ucf[(size_t)row * D_INNER + d];
    float du    = delta * u;
    float y     = 0.f;
#pragma unroll
    for (int s = 0; s < D_STATE; ++s) {
      float dA = __expf(delta * A[s]);
      h[s] = dA * h[s] + du * bc[cur][s];
      y   += h[s] * bc[cur][D_STATE + s];
    }
    float z = xz[(size_t)row * (2 * D_INNER) + D_INNER + d];
    float g = z / (1.f + __expf(-z));                      // SiLU gate
    ybf[(size_t)row * D_INNER + d] = f2bf((y + Dv * u) * g);
    __syncthreads();
  }
}

// ---------------------------------------------------------------------------
// Host-side orchestration
// ---------------------------------------------------------------------------
extern "C" void kernel_launch(void* const* d_in, const int* in_sizes, int n_in,
                              void* d_out, int out_size, void* d_ws, size_t ws_size,
                              hipStream_t stream) {
  (void)in_sizes; (void)n_in; (void)out_size; (void)ws_size;
  const float* x      = (const float*)d_in[0];
  const float* norm_w = (const float*)d_in[1];
  const float* norm_b = (const float*)d_in[2];
  const float* W_in   = (const float*)d_in[3];
  const float* conv_w = (const float*)d_in[4];
  const float* conv_b = (const float*)d_in[5];
  const float* W_x    = (const float*)d_in[6];
  const float* W_dt   = (const float*)d_in[7];
  const float* b_dt   = (const float*)d_in[8];
  const float* A_log  = (const float*)d_in[9];
  const float* Dp     = (const float*)d_in[10];
  const float* W_out  = (const float*)d_in[11];
  float* out = (float*)d_out;

  char* ws = (char*)d_ws;
  size_t off = 0;
  auto alloc = [&](size_t bytes) -> void* {
    void* p = ws + off;
    off += (bytes + 255) & ~(size_t)255;
    return p;
  };

  u16*   xn_bf   = (u16*)  alloc((size_t)NTOK * D_MODEL * 2);
  u16*   Win_bf  = (u16*)  alloc((size_t)2 * D_INNER * D_MODEL * 2);
  u16*   Wx_bf   = (u16*)  alloc((size_t)XPROJ_NPAD * D_INNER * 2);  // padded rows
  u16*   Wdt_bf  = (u16*)  alloc((size_t)D_INNER * DT_RANK * 2);
  u16*   Wout_bf = (u16*)  alloc((size_t)D_MODEL * D_INNER * 2);
  float* xz      = (float*)alloc((size_t)NTOK * 2 * D_INNER * 4);
  float* ucf     = (float*)alloc((size_t)NTOK * D_INNER * 4);
  u16*   ucb     = (u16*)  alloc((size_t)NTOK * D_INNER * 2);
  float* xdbl    = (float*)alloc((size_t)NTOK * XPROJ_N * 4);
  u16*   dt_bf   = (u16*)  alloc((size_t)NTOK * DT_RANK * 2);
  float* dpre    = (float*)alloc((size_t)NTOK * D_INNER * 4);
  u16*   ybf     = (u16*)  alloc((size_t)NTOK * D_INNER * 2);

  dim3 blk(256);
  auto cdiv = [](int a, int b) { return (a + b - 1) / b; };

  // weight down-conversions (independent)
  k_f32_to_bf16<<<cdiv(2 * D_INNER * D_MODEL, 256), blk, 0, stream>>>(
      W_in, Win_bf, 2 * D_INNER, 2 * D_INNER, D_MODEL, D_MODEL, D_MODEL);
  k_f32_to_bf16<<<cdiv(XPROJ_NPAD * D_INNER, 256), blk, 0, stream>>>(
      W_x, Wx_bf, XPROJ_N, XPROJ_NPAD, D_INNER, D_INNER, D_INNER);
  k_f32_to_bf16<<<cdiv(D_INNER * DT_RANK, 256), blk, 0, stream>>>(
      W_dt, Wdt_bf, D_INNER, D_INNER, DT_RANK, DT_RANK, DT_RANK);
  k_f32_to_bf16<<<cdiv(D_MODEL * D_INNER, 256), blk, 0, stream>>>(
      W_out, Wout_bf, D_MODEL, D_MODEL, D_INNER, D_INNER, D_INNER);

  // layernorm -> bf16 activations
  k_layernorm_bf16<<<NTOK, blk, 0, stream>>>(x, norm_w, norm_b, xn_bf);

  // in_proj: [4096,1024] x [4096,1024]^T -> xz[4096,4096]
  {
    dim3 g(cdiv(2 * D_INNER, BN), cdiv(NTOK, BM));
    k_gemm_bf16_nt<<<g, blk, 0, stream>>>(xn_bf, D_MODEL, Win_bf, D_MODEL,
                                          xz, 2 * D_INNER,
                                          NTOK, 2 * D_INNER, D_MODEL, nullptr);
  }

  // causal depthwise conv + SiLU
  k_conv_silu<<<cdiv(NTOK * D_INNER, 256), blk, 0, stream>>>(xz, conv_w, conv_b,
                                                             ucf, ucb);

  // x_proj: [4096,2048] x [96,2048]^T -> xdbl[4096,96]  (B padded to 128 rows)
  {
    dim3 g(cdiv(XPROJ_N, BN), cdiv(NTOK, BM));
    k_gemm_bf16_nt<<<g, blk, 0, stream>>>(ucb, D_INNER, Wx_bf, D_INNER,
                                          xdbl, XPROJ_N,
                                          NTOK, XPROJ_N, D_INNER, nullptr);
  }

  // dt slice -> bf16
  k_f32_to_bf16<<<cdiv(NTOK * DT_RANK, 256), blk, 0, stream>>>(
      xdbl, dt_bf, NTOK, NTOK, DT_RANK, XPROJ_N, DT_RANK);

  // dt_proj: [4096,64] x [2048,64]^T -> dpre[4096,2048]
  {
    dim3 g(cdiv(D_INNER, BN), cdiv(NTOK, BM));
    k_gemm_bf16_nt<<<g, blk, 0, stream>>>(dt_bf, DT_RANK, Wdt_bf, DT_RANK,
                                          dpre, D_INNER,
                                          NTOK, D_INNER, DT_RANK, nullptr);
  }

  // fused selective scan + gating -> ybf
  k_scan<<<(BATCH * D_INNER) / 256, blk, 0, stream>>>(dpre, xdbl, ucf, xz,
                                                      A_log, b_dt, Dp, ybf);

  // out_proj + residual: [4096,2048] x [1024,2048]^T + x -> out[4096,1024]
  {
    dim3 g(cdiv(D_MODEL, BN), cdiv(NTOK, BM));
    k_gemm_bf16_nt<<<g, blk, 0, stream>>>(ybf, D_INNER, Wout_bf, D_INNER,
                                          out, D_MODEL,
                                          NTOK, D_MODEL, D_INNER, x);
  }
}
